// ArcticExpertChoiceRouter_54211077210369
// MI455X (gfx1250) — matrix operand order repaired and checked by reading, not compiled, not run on hardware
//
#include <hip/hip_runtime.h>
#include <stdint.h>
#include <math.h>

#define T_TOK 16384
#define HID   4096
#define NEXP  128
#define CAP   160

typedef __attribute__((ext_vector_type(16))) __bf16 v16bf;
typedef __attribute__((ext_vector_type(2)))  __bf16 v2bf;
typedef __attribute__((ext_vector_type(8)))  float  v8f;

union FragBF { v16bf v; uint4 q[2]; };

#if __has_builtin(__builtin_amdgcn_global_load_async_to_lds_b128)
#define HAVE_ASYNC 1
#else
#define HAVE_ASYNC 0
#endif

// Builtin expects pointer-to-v4i in global (AS1) / LDS (AS3) address spaces.
typedef int v4i_vs __attribute__((vector_size(4 * sizeof(int))));
typedef v4i_vs GVec __attribute__((address_space(1)));
typedef v4i_vs LVec __attribute__((address_space(3)));

__device__ __forceinline__ void async_copy16(const void* g, const void* l) {
#if HAVE_ASYNC
  // global_load_async_to_lds_b128: per-lane 16B, tracked by ASYNCcnt
  __builtin_amdgcn_global_load_async_to_lds_b128(
      (GVec*)(uintptr_t)g,
      (LVec*)(unsigned int)(uintptr_t)l, 0, 0);
#else
  *(uint4*)(uintptr_t)l = *(const uint4*)(uintptr_t)g;
#endif
}

template <int N>
__device__ __forceinline__ void wait_async() {
#if HAVE_ASYNC
#if __has_builtin(__builtin_amdgcn_s_wait_asynccnt)
  __builtin_amdgcn_s_wait_asynccnt(N);
#else
  asm volatile("s_wait_asynccnt %0" ::"i"(N) : "memory");
#endif
#endif
}

// Native casts -> clang selects packed v_cvt_pk_bf16_f32 when available (RNE).
__device__ __forceinline__ unsigned int pack_bf16(float lo, float hi) {
  union { v2bf v; unsigned int u; } cv;
  cv.v = (v2bf){(__bf16)lo, (__bf16)hi};
  return cv.u;
}

// ---------------- Kernel 1: W fp32 -> bf16 (1 MB, stays hot in L2) ----------
__global__ void convW_kernel(const float* __restrict__ W,
                             unsigned short* __restrict__ Wb) {
  int i = (blockIdx.x * blockDim.x + threadIdx.x) * 8;
  float4 a = *(const float4*)(W + i);
  float4 b = *(const float4*)(W + i + 4);
  *(uint4*)(Wb + i) = make_uint4(pack_bf16(a.x, a.y), pack_bf16(a.z, a.w),
                                 pack_bf16(b.x, b.y), pack_bf16(b.z, b.w));
}

// ---------------- Kernel 2: logitsT[e][t] = x[t]·W[e] via bf16 WMMA ---------
// Block = 256 threads (8 wave32), tile = 128 tokens x 128 experts, K-step 32.
// Double-buffered LDS stages + register prefetch of next x chunk + async W
// DMA one chunk ahead: one barrier per k-step, HBM latency overlapped.
#define LROW 40  // LDS row stride in halves (80 B: 16B-aligned, bank-skewed)
#define KCH  (HID / 32)

__global__ void __launch_bounds__(256)
gemm_logits_kernel(const float* __restrict__ X,
                   const unsigned short* __restrict__ Wb,
                   float* __restrict__ logitsT) {
  __shared__ unsigned short xs[2][128 * LROW];
  __shared__ unsigned short ws[2][128 * LROW];

  const int tid  = threadIdx.x;
  const int lane = tid & 31;
  const int wv   = tid >> 5;        // wave id 0..7 -> 16-token stripe
  const int m16  = lane & 15;
  const int hig  = lane >> 4;       // half-wave group (ISA A/B striping)

  v8f acc[8];
#pragma unroll
  for (int i = 0; i < 8; ++i) acc[i] = (v8f){0.f,0.f,0.f,0.f,0.f,0.f,0.f,0.f};

  const int rowBase = blockIdx.x * 128;
  const int xrow  = tid >> 1;           // 0..127
  const int xhalf = (tid & 1) * 16;     // 16 floats per thread
  const float* xsrc = X + (size_t)(rowBase + xrow) * HID + xhalf;

  // W async-DMA source coordinates (2 x 16B per thread per chunk)
  const int we0 = tid >> 2,           wq0 = tid & 3;
  const int we1 = (tid + 256) >> 2,   wq1 = (tid + 256) & 3;

  // ---- prologue: stage chunk 0 ----
  async_copy16(Wb + (size_t)we0 * HID + wq0 * 8, &ws[0][we0 * LROW + wq0 * 8]);
  async_copy16(Wb + (size_t)we1 * HID + wq1 * 8, &ws[0][we1 * LROW + wq1 * 8]);
  float4 f0 = *(const float4*)(xsrc);
  float4 f1 = *(const float4*)(xsrc + 4);
  float4 f2 = *(const float4*)(xsrc + 8);
  float4 f3 = *(const float4*)(xsrc + 12);

  for (int kc = 0; kc < KCH; ++kc) {
    const int cur = kc & 1;

    // Convert prefetched x regs -> bf16 -> LDS (buffer cur)
    uint4 u0 = make_uint4(pack_bf16(f0.x, f0.y), pack_bf16(f0.z, f0.w),
                          pack_bf16(f1.x, f1.y), pack_bf16(f1.z, f1.w));
    uint4 u1 = make_uint4(pack_bf16(f2.x, f2.y), pack_bf16(f2.z, f2.w),
                          pack_bf16(f3.x, f3.y), pack_bf16(f3.z, f3.w));
    *(uint4*)&xs[cur][xrow * LROW + xhalf]     = u0;
    *(uint4*)&xs[cur][xrow * LROW + xhalf + 8] = u1;

    if (kc + 1 < KCH) {
      // Prefetch next chunk: W via async DMA into other buffer, x into regs.
      const int k1 = (kc + 1) * 32;
      async_copy16(Wb + (size_t)we0 * HID + k1 + wq0 * 8,
                   &ws[cur ^ 1][we0 * LROW + wq0 * 8]);
      async_copy16(Wb + (size_t)we1 * HID + k1 + wq1 * 8,
                   &ws[cur ^ 1][we1 * LROW + wq1 * 8]);
      f0 = *(const float4*)(xsrc + k1);
      f1 = *(const float4*)(xsrc + k1 + 4);
      f2 = *(const float4*)(xsrc + k1 + 8);
      f3 = *(const float4*)(xsrc + k1 + 12);
      wait_async<2>();   // chunk kc's 2 DMAs done; next chunk stays in flight
    } else {
      wait_async<0>();
    }
    __syncthreads();     // xs/ws[cur] ready for everyone

    // A fragment (16x32 bf16: lanes 0-15 K{0-7,16-23}, 16-31 K{8-15,24-31})
    FragBF a;
    {
      const int aoff = (wv * 16 + m16) * LROW + hig * 8;
      a.q[0] = *(const uint4*)&xs[cur][aoff];
      a.q[1] = *(const uint4*)&xs[cur][aoff + 16];
    }
    // 8 N-tiles: B fragment (32x16: lanes 0-15 K{0-7,16-23}, 16-31 K{8-15,24-31})
#pragma unroll
    for (int nt = 0; nt < 8; ++nt) {
      FragBF b;
      const int boff = (nt * 16 + m16) * LROW + hig * 8;
      b.q[0] = *(const uint4*)&ws[cur][boff];
      b.q[1] = *(const uint4*)&ws[cur][boff + 16];
      acc[nt] = __builtin_amdgcn_wmma_f32_16x16x32_bf16(
          false, a.v, false, b.v, (short)0, acc[nt], false, false);
    }
    // No trailing barrier: next iteration writes the other xs buffer, but DMAs W into ws[cur].
  }

  // D layout: VGPR r -> M = (lane>=16?8:0)+r ; N = lane&15. Store transposed [E][T].
  const int tokBase = rowBase + wv * 16 + hig * 8;
#pragma unroll
  for (int nt = 0; nt < 8; ++nt) {
    const int e = nt * 16 + m16;
    float* dst = logitsT + (size_t)e * T_TOK + tokBase;
#pragma unroll
    for (int r = 0; r < 8; ++r) dst[r] = acc[nt][r];
  }
}

// ---------------- Kernel 3: zero dispatch mask ------------------------------
__global__ void zero_kernel(float4* __restrict__ p, int n4) {
  int i = blockIdx.x * blockDim.x + threadIdx.x;
  if (i < n4) p[i] = make_float4(0.f, 0.f, 0.f, 0.f);
}

// ---------------- Kernel 4: per-expert exact top-160 ------------------------
// One block per expert. 4-pass radix select on order-preserving uint keys,
// then bitonic sort of winners (desc value, asc index tie-break like top_k).
__device__ __forceinline__ unsigned int fkey(float f) {
  unsigned int u = __float_as_uint(f);
  return (u & 0x80000000u) ? ~u : (u | 0x80000000u);
}

__global__ void __launch_bounds__(256)
topk_expert_kernel(const float* __restrict__ logitsT,
                   float* __restrict__ outIdx,
                   float* __restrict__ outMask,
                   float* __restrict__ counts) {
  __shared__ unsigned int hist[256];
  __shared__ unsigned long long keys[256];
  __shared__ unsigned int scnt;

  const int e   = blockIdx.x;
  const int tid = threadIdx.x;
  const float* col = logitsT + (size_t)e * T_TOK;

  unsigned int prefix = 0, pmask = 0;
  int krem = CAP;
  for (int shift = 24; shift >= 0; shift -= 8) {
    hist[tid] = 0;
    __syncthreads();
    for (int t = tid; t < T_TOK; t += 256) {
      unsigned int u = fkey(col[t]);
      if ((u & pmask) == prefix) atomicAdd(&hist[(u >> shift) & 255u], 1u);
    }
    __syncthreads();
    unsigned int c = 0; int b = 0;       // uniform scan (same result all threads)
    for (int i = 255; i >= 0; --i) {
      unsigned int h = hist[i];
      if (c + h >= (unsigned int)krem) { b = i; break; }
      c += h;
    }
    krem   -= (int)c;
    prefix |= ((unsigned int)b) << shift;
    pmask  |= 0xFFu << shift;
    __syncthreads();
  }
  const unsigned int Kth = prefix;       // exact 160th-largest key

  if (tid == 0) scnt = 0;
  __syncthreads();
  for (int t = tid; t < T_TOK; t += 256) {
    unsigned int u = fkey(col[t]);
    if (u >= Kth) {
      unsigned int s = atomicAdd(&scnt, 1u);
      if (s < 256u)
        keys[s] = ((unsigned long long)u << 32) | (unsigned int)~t;  // ~t: asc-index ties
    }
  }
  __syncthreads();
  unsigned int nsel = scnt; if (nsel > 256u) nsel = 256u;
  if ((unsigned int)tid >= nsel) keys[tid] = 0ull;
  __syncthreads();

  // bitonic sort, descending
  for (unsigned int k = 2; k <= 256; k <<= 1) {
    for (unsigned int j = k >> 1; j > 0; j >>= 1) {
      unsigned int ixj = (unsigned int)tid ^ j;
      if (ixj > (unsigned int)tid) {
        unsigned long long A = keys[tid], B = keys[ixj];
        bool desc = (((unsigned int)tid & k) == 0);
        if ((A < B) == desc) { keys[tid] = B; keys[ixj] = A; }
      }
      __syncthreads();
    }
  }

  if (tid < CAP) {
    unsigned int tok = ~(unsigned int)(keys[tid] & 0xFFFFFFFFull);
    outIdx[e * CAP + tid] = (float)tok;                 // indices as f32 output
    outMask[(size_t)tok * NEXP + e] = 1.0f;             // scatter dispatch mask
  }
  if (tid == 0) counts[e] = (float)CAP;
}

// ---------------- Kernel 5: load-balancing loss -----------------------------
__global__ void loss_kernel(const float* __restrict__ counts,
                            float* __restrict__ outLoss) {
  __shared__ float sh[NEXP];
  int tid = threadIdx.x;
  float v = counts[tid];
  sh[tid] = v;
  __syncthreads();
  for (int s = NEXP / 2; s > 0; s >>= 1) {
    if (tid < s) sh[tid] += sh[tid + s];
    __syncthreads();
  }
  float mean = sh[0] / (float)NEXP;
  __syncthreads();
  sh[tid] = v * logf(v / mean);
  __syncthreads();
  for (int s = NEXP / 2; s > 0; s >>= 1) {
    if (tid < s) sh[tid] += sh[tid + s];
    __syncthreads();
  }
  if (tid == 0) *outLoss = sh[0] / (float)NEXP;
}

// ---------------- Host launcher --------------------------------------------
extern "C" void kernel_launch(void* const* d_in, const int* in_sizes, int n_in,
                              void* d_out, int out_size, void* d_ws, size_t ws_size,
                              hipStream_t stream) {
  const float* x = (const float*)d_in[0];   // [T, H] fp32
  const float* W = (const float*)d_in[1];   // [E, H] fp32

  float* out     = (float*)d_out;
  float* outIdx  = out;                                          // E*CAP
  float* outMask = out + (size_t)NEXP * CAP;                     // T*E
  float* outLoss = out + (size_t)NEXP * CAP + (size_t)T_TOK * NEXP;

  char* ws = (char*)d_ws;
  unsigned short* Wb = (unsigned short*)ws;                       // 1 MB bf16 W
  float* logitsT = (float*)(ws + (size_t)NEXP * HID * 2);         // 8 MB [E][T]
  float* counts  = (float*)(ws + (size_t)NEXP * HID * 2
                               + (size_t)NEXP * T_TOK * 4);       // 512 B

  convW_kernel<<<(NEXP * HID / 8) / 256, 256, 0, stream>>>(W, Wb);
  gemm_logits_kernel<<<T_TOK / 128, 256, 0, stream>>>(x, Wb, logitsT);
  zero_kernel<<<(T_TOK * NEXP / 4) / 256, 256, 0, stream>>>((float4*)outMask,
                                                            T_TOK * NEXP / 4);
  topk_expert_kernel<<<NEXP, 256, 0, stream>>>(logitsT, outIdx, outMask, counts);
  loss_kernel<<<1, NEXP, 0, stream>>>(counts, outLoss);
}
